// ImprovedHeteroGNN_2138893713892
// MI455X (gfx1250) — matrix operand order repaired and checked by reading, not compiled, not run on hardware
//
#include <hip/hip_runtime.h>
#include <hip/hip_bf16.h>

#define H      128
#define OUTC   64
#define LN_EPS 1e-5f
#define FSTR   132   // f32 LDS row stride: 16B-aligned rows, float4-friendly
#define BSTR   136   // bf16 LDS row stride: 272B rows -> 16B aligned, banks staggered

typedef __attribute__((ext_vector_type(16))) __bf16 v16bf;
typedef __attribute__((ext_vector_type(8)))  __bf16 v8bf;
typedef __attribute__((ext_vector_type(4)))  __bf16 v4bf;
typedef __attribute__((ext_vector_type(8)))  float  v8f;

// ---------------------------------------------------------------------------
// CDNA5 WMMA fragment layouts (wave32):
// A (16x32 bf16): lane m=lane&15 is row M; half h=lane>>4; element e:
//   K = kk*32 + h*16 + e   -> one contiguous 16-elem run (this code's order, same as B)
// B (32x16 bf16): lane n=lane&15 is col N; element e: K = kk*32 + h*16 + e (this code's order; AMD's table order differs)
// C/D (16x16 f32): VGPR v holds M = v + 8*h, N = lane&15
// ---------------------------------------------------------------------------
__device__ __forceinline__ v16bf load_a_frag(const __bf16* __restrict__ lds,
                                             int kk, int lane) {
  const int m = lane & 15, h = lane >> 4;
  const __bf16* p = lds + m * BSTR + kk * 32 + h * 16;
  const v8bf lo = *(const v8bf*)p;          // ds_load_b128
  const v8bf hi = *(const v8bf*)(p + 8);    // ds_load_b128
  return __builtin_shufflevector(lo, hi, 0, 1, 2, 3, 4, 5, 6, 7,
                                 8, 9, 10, 11, 12, 13, 14, 15);
}

__device__ __forceinline__ v16bf load_b_frag(const float* __restrict__ wrow,
                                             int kk, int lane) {
  const int h = (lane >> 4) & 1;
  const float* p = wrow + kk * 32 + h * 16;  // 16 consecutive f32 (global b128 x4)
  v16bf b;
#pragma unroll
  for (int e = 0; e < 16; ++e) b[e] = (__bf16)p[e];
  return b;
}

__device__ __forceinline__ v8f wmma_bf16(v16bf a, v16bf b, v8f c) {
  return __builtin_amdgcn_wmma_f32_16x16x32_bf16(false, a, false, b, (short)0, c,
                                                 false, false);
}

// ---------------------------------------------------------------------------
__global__ void zero_f32(float* __restrict__ p, long n) {
  long i = (long)blockIdx.x * blockDim.x + threadIdx.x;
  const long stride = (long)gridDim.x * blockDim.x;
  for (; i < n; i += stride) p[i] = 0.0f;
}

// Edge scatter: agg[dst] += x_src[src]; deg[dst] += 1 (L2-resident buffers).
__global__ __launch_bounds__(256)
void scatter_add(const float* __restrict__ xsrc, const int* __restrict__ ei,
                 float* __restrict__ agg, float* __restrict__ deg, int E) {
  const long t = (long)blockIdx.x * blockDim.x + threadIdx.x;
  if (t >= (long)E * 32) return;
  const int e   = (int)(t >> 5);
  const int c4  = ((int)t & 31) << 2;
  const int src = ei[e];       // ei[0] row
  const int dst = ei[E + e];   // ei[1] row
  const float4 v = *(const float4*)(xsrc + (long)src * H + c4);
  float* a = agg + (long)dst * H + c4;
  atomicAdd(a + 0, v.x);
  atomicAdd(a + 1, v.y);
  atomicAdd(a + 2, v.z);
  atomicAdd(a + 3, v.w);
  if ((t & 31) == 0) atomicAdd(deg + dst, 1.0f);
}

// ---------------------------------------------------------------------------
// Fused SAGE combine, 32 rows/block (two 16-row tiles), weight fragments held
// in registers across tiles. 8 waves; wave w owns output-column tile w.
// ---------------------------------------------------------------------------
template <int NSAGE, bool FINAL>
__global__ __launch_bounds__(256)
void hetero_combine(const float* __restrict__ agg0, const float* __restrict__ deg0,
                    const float* __restrict__ Wl0,  const float* __restrict__ bl0,
                    const float* __restrict__ Wr0,
                    const float* __restrict__ agg1, const float* __restrict__ deg1,
                    const float* __restrict__ Wl1,  const float* __restrict__ bl1,
                    const float* __restrict__ Wr1,
                    const float* __restrict__ xdst,
                    const float* __restrict__ gamma, const float* __restrict__ beta,
                    float* __restrict__ hout,
                    const float* __restrict__ Wout, const float* __restrict__ bout,
                    float* __restrict__ outp) {
  __shared__ __align__(16) __bf16 s_a0[16 * BSTR];
  __shared__ __align__(16) __bf16 s_a1[(NSAGE == 2) ? 16 * BSTR : 8];
  __shared__ __align__(16) __bf16 s_rt[16 * BSTR];
  __shared__ __align__(16) float  s_rootf[16 * FSTR];
  __shared__ __align__(16) float  s_pre[16 * FSTR];
  __shared__ float s_scl0[16], s_scl1[16], s_mu[16], s_rs[16];

  const int tid  = threadIdx.x;
  const int wave = tid >> 5, lane = tid & 31;
  const int n    = wave * 16 + (lane & 15);  // this wave's output channel
  const int hh   = lane >> 4;

  // ---- Preload weight B-fragments once per block (reused by both row tiles)
  v16bf bl_f[4], br_f[4], bl1_f[4], br1_f[4];
  {
    const float* wl0row = Wl0 + n * H;       // B[k][n] = W[n][k]
    const float* wr0row = Wr0 + n * H;
    __builtin_prefetch(wl0row, 0, 0);        // global_prefetch
    __builtin_prefetch(wr0row, 0, 0);
#pragma unroll
    for (int kk = 0; kk < 4; ++kk) {
      bl_f[kk] = load_b_frag(wl0row, kk, lane);
      br_f[kk] = load_b_frag(wr0row, kk, lane);
    }
    if (NSAGE == 2) {
      const float* wl1row = Wl1 + n * H;
      const float* wr1row = Wr1 + n * H;
#pragma unroll
      for (int kk = 0; kk < 4; ++kk) {
        bl1_f[kk] = load_b_frag(wl1row, kk, lane);
        br1_f[kk] = load_b_frag(wr1row, kk, lane);
      }
    }
  }
  const float scale = (NSAGE == 2) ? 0.5f : 1.0f;
  const float bias  = (NSAGE == 2) ? 0.5f * (bl0[n] + bl1[n]) : bl0[n];

  for (int t = 0; t < 2; ++t) {
    const int row0 = blockIdx.x * 32 + t * 16;
    if (t) __syncthreads();  // previous tile done with LDS before restaging

    if (tid < 16) {
      s_scl0[tid] = 1.0f / fmaxf(deg0[row0 + tid], 1.0f);
      if (NSAGE == 2) s_scl1[tid] = 1.0f / fmaxf(deg1[row0 + tid], 1.0f);
    }
    __syncthreads();

    // ---- Stage tiles: convert fp32 -> bf16 ONCE, store in fragment layout.
    for (int i = tid; i < 512; i += 256) {
      const int r = i >> 5, c = (i & 31) << 2;
      const long g = (long)(row0 + r) * H + c;

      const float4 va = *(const float4*)(agg0 + g);
      const float s0 = s_scl0[r];
      v4bf pa;
      pa[0] = (__bf16)(va.x * s0); pa[1] = (__bf16)(va.y * s0);
      pa[2] = (__bf16)(va.z * s0); pa[3] = (__bf16)(va.w * s0);
      *(v4bf*)(s_a0 + r * BSTR + c) = pa;

      if (NSAGE == 2) {
        const float4 vb = *(const float4*)(agg1 + g);
        const float s1 = s_scl1[r];
        v4bf pb;
        pb[0] = (__bf16)(vb.x * s1); pb[1] = (__bf16)(vb.y * s1);
        pb[2] = (__bf16)(vb.z * s1); pb[3] = (__bf16)(vb.w * s1);
        *(v4bf*)(s_a1 + r * BSTR + c) = pb;
      }

      const float4 vr = *(const float4*)(xdst + g);
      v4bf pr;
      pr[0] = (__bf16)vr.x; pr[1] = (__bf16)vr.y;
      pr[2] = (__bf16)vr.z; pr[3] = (__bf16)vr.w;
      *(v4bf*)(s_rt + r * BSTR + c) = pr;
      *(float4*)(s_rootf + r * FSTR + c) = vr;  // fp32 copy for the residual
    }
    __syncthreads();

    // ---- GEMM: b128 LDS fragment loads feeding bf16 WMMA
    v8f acc = {};
#pragma unroll
    for (int kk = 0; kk < 4; ++kk) {
      acc = wmma_bf16(load_a_frag(s_a0, kk, lane), bl_f[kk], acc);
      acc = wmma_bf16(load_a_frag(s_rt, kk, lane), br_f[kk], acc);
    }
    if (NSAGE == 2) {
#pragma unroll
      for (int kk = 0; kk < 4; ++kk) {
        acc = wmma_bf16(load_a_frag(s_a1, kk, lane), bl1_f[kk], acc);
        acc = wmma_bf16(load_a_frag(s_rt, kk, lane), br1_f[kk], acc);
      }
    }
#pragma unroll
    for (int v = 0; v < 8; ++v)
      s_pre[(v + 8 * hh) * FSTR + n] = acc[v] * scale + bias;
    __syncthreads();

    // ---- LayerNorm stats (fp32 accumulator precision)
    if (tid < 16) {
      const float* row = s_pre + tid * FSTR;
      float sum = 0.0f;
      for (int c = 0; c < H; c += 4) {
        const float4 v = *(const float4*)(row + c);
        sum += v.x + v.y + v.z + v.w;
      }
      const float mu = sum * (1.0f / H);
      float var = 0.0f;
      for (int c = 0; c < H; c += 4) {
        const float4 v = *(const float4*)(row + c);
        const float dx = v.x - mu, dy = v.y - mu, dz = v.z - mu, dw = v.w - mu;
        var += dx * dx + dy * dy + dz * dz + dw * dw;
      }
      s_mu[tid] = mu;
      s_rs[tid] = rsqrtf(var * (1.0f / H) + LN_EPS);
    }
    __syncthreads();

    if (!FINAL) {
      for (int i = tid; i < 512; i += 256) {
        const int r = i >> 5, c = (i & 31) << 2;
        const float4 p  = *(const float4*)(s_pre + r * FSTR + c);
        const float4 rt = *(const float4*)(s_rootf + r * FSTR + c);
        const float4 gm = *(const float4*)(gamma + c);
        const float4 bt = *(const float4*)(beta + c);
        const float mu = s_mu[r], rs = s_rs[r];
        float4 y;
        y.x = fmaxf((p.x - mu) * rs * gm.x + bt.x, 0.0f) + rt.x;
        y.y = fmaxf((p.y - mu) * rs * gm.y + bt.y, 0.0f) + rt.y;
        y.z = fmaxf((p.z - mu) * rs * gm.z + bt.z, 0.0f) + rt.z;
        y.w = fmaxf((p.w - mu) * rs * gm.w + bt.w, 0.0f) + rt.w;
        *(float4*)(hout + (long)(row0 + r) * H + c) = y;
      }
    } else {
      // new_s -> bf16 fragment layout in s_a0 (reuse), then fused 128->64 GEMM.
      for (int i = tid; i < 512; i += 256) {
        const int r = i >> 5, c = (i & 31) << 2;
        const float4 p  = *(const float4*)(s_pre + r * FSTR + c);
        const float4 rt = *(const float4*)(s_rootf + r * FSTR + c);
        const float4 gm = *(const float4*)(gamma + c);
        const float4 bt = *(const float4*)(beta + c);
        const float mu = s_mu[r], rs = s_rs[r];
        v4bf py;
        py[0] = (__bf16)(fmaxf((p.x - mu) * rs * gm.x + bt.x, 0.0f) + rt.x);
        py[1] = (__bf16)(fmaxf((p.y - mu) * rs * gm.y + bt.y, 0.0f) + rt.y);
        py[2] = (__bf16)(fmaxf((p.z - mu) * rs * gm.z + bt.z, 0.0f) + rt.z);
        py[3] = (__bf16)(fmaxf((p.w - mu) * rs * gm.w + bt.w, 0.0f) + rt.w);
        *(v4bf*)(s_a0 + r * BSTR + c) = py;
      }
      __syncthreads();
      if (wave < 4) {  // wave-uniform: EXEC stays all-ones for WMMA
        const int nf = wave * 16 + (lane & 15);
        const float* worow = Wout + nf * H;
        v8f o = {};
#pragma unroll
        for (int kk = 0; kk < 4; ++kk)
          o = wmma_bf16(load_a_frag(s_a0, kk, lane),
                        load_b_frag(worow, kk, lane), o);
        const float bo = bout[nf];
#pragma unroll
        for (int v = 0; v < 8; ++v)
          outp[(long)(row0 + v + 8 * hh) * OUTC + nf] = o[v] + bo;
      }
    }
  }
}

// ---------------------------------------------------------------------------
extern "C" void kernel_launch(void* const* d_in, const int* in_sizes, int n_in,
                              void* d_out, int out_size, void* d_ws, size_t ws_size,
                              hipStream_t stream) {
  const float* x_s  = (const float*)d_in[0];
  const float* x_o  = (const float*)d_in[1];
  const float* x_f  = (const float*)d_in[2];
  const int*   ei0  = (const int*)d_in[3];
  const int*   ei1  = (const int*)d_in[4];
  const int*   ei2  = (const int*)d_in[5];
  const int*   ei3  = (const int*)d_in[6];
  const float* Wl1  = (const float*)d_in[7];
  const float* bl1  = (const float*)d_in[8];
  const float* Wr1  = (const float*)d_in[9];
  const float* Wl2  = (const float*)d_in[10];
  const float* bl2  = (const float*)d_in[11];
  const float* Wr2  = (const float*)d_in[12];
  const float* g    = (const float*)d_in[13];
  const float* bta  = (const float*)d_in[14];
  const float* Wout = (const float*)d_in[15];
  const float* bout = (const float*)d_in[16];
  float* out = (float*)d_out;

  const int  N  = in_sizes[0] / H;   // 100000 (divisible by 32)
  const int  E  = in_sizes[3] / 2;   // 600000
  const long NH = (long)N * H;
  const int  HH = H * H;

  float* ws   = (float*)d_ws;
  float* h_s  = ws;            // layer-1 study output
  float* h_o  = ws + NH;       // layer-1 outcome output
  float* h_f  = ws + 2 * NH;   // layer-1 facility output
  float* aggA = ws + 3 * NH;   // agg buffer A
  float* aggB = ws + 4 * NH;   // agg buffer B (contiguous with A)
  float* degA = ws + 5 * NH;
  float* degB = degA + N;      // contiguous with degA

  const dim3 cgrid(N / 32), cblk(256);
  const int  sc_blocks = (int)(((long)E * 32 + 255) / 256);
  const int  ZB = 2048, ZT = 256;

  // ================= Layer 1 =================
  // edge type 0: study -> outcome
  zero_f32<<<ZB, ZT, 0, stream>>>(aggA, NH);
  zero_f32<<<64, ZT, 0, stream>>>(degA, N);
  scatter_add<<<sc_blocks, 256, 0, stream>>>(x_s, ei0, aggA, degA, E);
  hetero_combine<1, false><<<cgrid, cblk, 0, stream>>>(
      aggA, degA, Wl1 + 0 * HH, bl1 + 0 * H, Wr1 + 0 * HH,
      nullptr, nullptr, nullptr, nullptr, nullptr,
      x_o, g, bta, h_o, nullptr, nullptr, nullptr);

  // edge type 2: study -> facility
  zero_f32<<<ZB, ZT, 0, stream>>>(aggA, NH);
  zero_f32<<<64, ZT, 0, stream>>>(degA, N);
  scatter_add<<<sc_blocks, 256, 0, stream>>>(x_s, ei2, aggA, degA, E);
  hetero_combine<1, false><<<cgrid, cblk, 0, stream>>>(
      aggA, degA, Wl1 + 2 * HH, bl1 + 2 * H, Wr1 + 2 * HH,
      nullptr, nullptr, nullptr, nullptr, nullptr,
      x_f, g, bta, h_f, nullptr, nullptr, nullptr);

  // edge types 1 (outcome->study) + 3 (facility->study), mean of the two
  zero_f32<<<ZB, ZT, 0, stream>>>(aggA, 2 * NH);
  zero_f32<<<64, ZT, 0, stream>>>(degA, 2 * (long)N);
  scatter_add<<<sc_blocks, 256, 0, stream>>>(x_o, ei1, aggA, degA, E);
  scatter_add<<<sc_blocks, 256, 0, stream>>>(x_f, ei3, aggB, degB, E);
  hetero_combine<2, false><<<cgrid, cblk, 0, stream>>>(
      aggA, degA, Wl1 + 1 * HH, bl1 + 1 * H, Wr1 + 1 * HH,
      aggB, degB, Wl1 + 3 * HH, bl1 + 3 * H, Wr1 + 3 * HH,
      x_s, g, bta, h_s, nullptr, nullptr, nullptr);

  // ================= Layer 2 (only study feeds the output head) ==============
  zero_f32<<<ZB, ZT, 0, stream>>>(aggA, 2 * NH);
  zero_f32<<<64, ZT, 0, stream>>>(degA, 2 * (long)N);
  scatter_add<<<sc_blocks, 256, 0, stream>>>(h_o, ei1, aggA, degA, E);
  scatter_add<<<sc_blocks, 256, 0, stream>>>(h_f, ei3, aggB, degB, E);
  hetero_combine<2, true><<<cgrid, cblk, 0, stream>>>(
      aggA, degA, Wl2 + 1 * HH, bl2 + 1 * H, Wr2 + 1 * HH,
      aggB, degB, Wl2 + 3 * HH, bl2 + 3 * H, Wr2 + 3 * HH,
      h_s, g + H, bta + H, nullptr, Wout, bout, out);

  (void)n_in; (void)out_size; (void)ws_size;
}
